// MixedSparseSingleLayerWithGate_26912265077367
// MI455X (gfx1250) — matrix operand order repaired and checked by reading, not compiled, not run on hardware
//
#include <hip/hip_runtime.h>

// ---------------- problem constants ----------------
#define S_LEN  2048
#define DMODEL 1024
#define FFN    2816
#define NHEAD  16
#define DHEAD  64
#define LRANK  16

typedef __attribute__((ext_vector_type(16))) _Float16 v16h;
typedef __attribute__((ext_vector_type(8)))  _Float16 v8h;
typedef __attribute__((ext_vector_type(8)))  float    v8f;

__device__ __constant__ float NF4_LUT[16] = {
    -1.0f, -0.6961928009986877f, -0.5250730514526367f, -0.39491748809814453f,
    -0.28444138169288635f, -0.18477343022823334f, -0.09105003625154495f, 0.0f,
    0.07958029955625534f, 0.16093020141124725f, 0.24611230194568634f, 0.33791524171829224f,
    0.44070982933044434f, 0.5626170039176941f, 0.7229568362236023f, 1.0f};

// ---- WMMA fragment helpers (ISA 7.12.2 layouts, wave32) ----
// A (16x32 f16): lane holds row M=lane%16; halves 0..7 = K(base..base+7),
// 8..15 = K(base+16..base+23), base = (lane<16 ? 0 : 8).
__device__ __forceinline__ v16h frag_a(const _Float16* p) {
  v8h lo = *(const v8h*)(p);
  v8h hi = *(const v8h*)(p + 16);
  v16h a;
#pragma unroll
  for (int i = 0; i < 8; ++i) { a[i] = lo[i]; a[i + 8] = hi[i]; }
  return a;
}
// B (32x16 f16): lane holds column N=lane%16; halves 0..7 = K(base..base+7),
// 8..15 = K(base+16..base+23), base = (lane<16 ? 0 : 8).  p points at B[base][n] stored [n][k].
__device__ __forceinline__ v16h frag_b(const _Float16* p) {
  v8h lo = *(const v8h*)(p);
  v8h hi = *(const v8h*)(p + 16);
  v16h b;
#pragma unroll
  for (int i = 0; i < 8; ++i) { b[i] = lo[i]; b[i + 8] = hi[i]; }
  return b;
}
__device__ __forceinline__ v8f wmma16(v16h a, v16h b, v8f c) {
  return __builtin_amdgcn_wmma_f32_16x16x32_f16(false, a, false, b, (short)0, c,
                                                false, false);
}

// ---------------- RMSNorm: f32 in -> f16 out ----------------
__launch_bounds__(256)
__global__ void rmsnorm_f16(const float* __restrict__ x, const float* __restrict__ w,
                            _Float16* __restrict__ out, int ncols) {
  const int row = blockIdx.x, tid = threadIdx.x;
  const float* xr = x + (size_t)row * ncols;
  float ss = 0.f;
  for (int i = tid; i < ncols; i += 256) { float v = xr[i]; ss += v * v; }
  __shared__ float red[256];
  red[tid] = ss; __syncthreads();
  for (int s = 128; s > 0; s >>= 1) {
    if (tid < s) red[tid] += red[tid + s];
    __syncthreads();
  }
  const float scale = rsqrtf(red[0] / (float)ncols + 1e-10f);
  for (int i = tid; i < ncols; i += 256)
    out[(size_t)row * ncols + i] = (_Float16)(xr[i] * scale * w[i]);
}

// ---------------- LoRA XA = X(f16) @ A(f32, IxR) -> f16 [S][16] ----------------
__launch_bounds__(256)
__global__ void lora_xa(const _Float16* __restrict__ X, const float* __restrict__ A,
                        _Float16* __restrict__ XA, int I) {
  const int s = blockIdx.x, tid = threadIdx.x;
  const int r = tid & 15, g = tid >> 4;  // 16 reduction groups x 16 ranks
  const _Float16* xr = X + (size_t)s * I;
  float acc = 0.f;
  for (int i = g; i < I; i += 16) acc += (float)xr[i] * A[i * LRANK + r];
  __shared__ float red[256];
  red[tid] = acc; __syncthreads();
  if (g == 0) {
    float t = 0.f;
#pragma unroll
    for (int gg = 0; gg < 16; ++gg) t += red[gg * 16 + r];
    XA[s * LRANK + r] = (_Float16)t;
  }
}

// ---------------- Fused NF4-dequant GEMM + LoRA-B + residual ----------------
// out[S][O] = X[S][I](f16) @ dequant(codes,absmax)[O][I]^T + XA[S][16] @ lb[16][O] (+res)
// X tile is staged with CDNA5 async global->LDS DMA (ASYNCcnt); the NF4 W tile
// is decoded on the VALU (LUT in LDS) since dequant must pass through registers.
#define LSTR 40  // padded LDS row stride in halfs (80B, 16B-aligned)
#define GEMM_LDS_BYTES (2 * 128 * LSTR * 2 + 64)
__launch_bounds__(256)
__global__ void gemm_nf4_lora(const _Float16* __restrict__ X,
                              const int* __restrict__ codes,
                              const float* __restrict__ absmax,
                              const _Float16* __restrict__ XA,
                              const float* __restrict__ lorab,
                              const float* __restrict__ res,
                              float* __restrict__ out, int I, int O) {
  extern __shared__ _Float16 smem[];            // dynamic LDS: offsets start at 0
  _Float16* Xs = smem;                          // [128][LSTR]  A tiles: [m][k]
  _Float16* Ws = smem + 128 * LSTR;             // [128][LSTR]  B tiles: [n][k]
  float* lut = (float*)(smem + 2 * 128 * LSTR); // 16 floats
  const int t = threadIdx.x;
  if (t < 16) lut[t] = NF4_LUT[t];
  const int lane = t & 31, lane16 = lane & 15, laneHalf = lane >> 4;
  const int wave = t >> 5, waveM = wave >> 2, waveN = wave & 3;
  const int blockN = blockIdx.x * 128, blockM = blockIdx.y * 128;
  const int cr = t >> 1;           // cooperative row 0..127
  const int cs = (t & 1) * 16;     // 0 or 16 (half of 32-wide k slab)
  const unsigned xsLds = (unsigned)((cr * LSTR + cs) * 2);  // LDS byte addr of this
                                                            // thread's X segment

  v8f acc[4][2];
#pragma unroll
  for (int mt = 0; mt < 4; ++mt)
#pragma unroll
    for (int nt = 0; nt < 2; ++nt) acc[mt][nt] = (v8f){};
  __syncthreads();

  for (int k0 = 0; k0 < I; k0 += 32) {
    // ---- stage X tile (128 x 32 halfs) via async global->LDS DMA ----
    // INST_OFFSET is added to BOTH the LDS and memory addresses (ISA 10.x),
    // so one address VGPR + offset:16 covers the second 16B.
    const _Float16* xg = X + (size_t)(blockM + cr) * I + k0 + cs;
    asm volatile("global_load_async_to_lds_b128 %0, %1, off\n\t"
                 "global_load_async_to_lds_b128 %0, %1, off offset:16"
                 :: "v"(xsLds), "v"(xg) : "memory");
    // ---- dequant W tile (128 n-rows x 32 k) on the VALU ----
    const size_t wbase = (size_t)(blockN + cr) * I + k0 + cs;
    const int* cg = codes + wbase;
    __builtin_prefetch(cg + 32, 0, 1);          // global_prefetch_b8
    const float am = absmax[wbase >> 6];        // 64-elem blocks, 16-aligned span
#pragma unroll
    for (int j = 0; j < 16; j += 4) {
      int4 c4 = *(const int4*)(cg + j);
      Ws[cr * LSTR + cs + j + 0] = (_Float16)(lut[c4.x & 15] * am);
      Ws[cr * LSTR + cs + j + 1] = (_Float16)(lut[c4.y & 15] * am);
      Ws[cr * LSTR + cs + j + 2] = (_Float16)(lut[c4.z & 15] * am);
      Ws[cr * LSTR + cs + j + 3] = (_Float16)(lut[c4.w & 15] * am);
    }
    asm volatile("s_wait_asynccnt 0" ::: "memory");  // async X data landed in LDS
    __syncthreads();
    // ---- 8 WMMAs per wave ----
    v16h a[4], b[2];
#pragma unroll
    for (int mt = 0; mt < 4; ++mt)
      a[mt] = frag_a(&Xs[(waveM * 64 + mt * 16 + lane16) * LSTR + laneHalf * 8]);
#pragma unroll
    for (int nt = 0; nt < 2; ++nt)
      b[nt] = frag_b(&Ws[(waveN * 32 + nt * 16 + lane16) * LSTR + laneHalf * 8]);
#pragma unroll
    for (int mt = 0; mt < 4; ++mt)
#pragma unroll
      for (int nt = 0; nt < 2; ++nt) acc[mt][nt] = wmma16(a[mt], b[nt], acc[mt][nt]);
    __syncthreads();
  }

  // ---- LoRA-B step: one K=32 slab, upper half zero-padded ----
  {
    int4 z4; z4.x = 0; z4.y = 0; z4.z = 0; z4.w = 0;
    if (cs == 0) {
      *(int4*)&Xs[cr * LSTR + 0] = *(const int4*)&XA[(blockM + cr) * LRANK];
      *(int4*)&Xs[cr * LSTR + 8] = *(const int4*)&XA[(blockM + cr) * LRANK + 8];
#pragma unroll
      for (int j = 0; j < 16; ++j)
        Ws[cr * LSTR + j] = (_Float16)lorab[j * O + blockN + cr];
    } else {
      *(int4*)&Xs[cr * LSTR + 16] = z4;
      *(int4*)&Xs[cr * LSTR + 24] = z4;
      *(int4*)&Ws[cr * LSTR + 16] = z4;
      *(int4*)&Ws[cr * LSTR + 24] = z4;
    }
    __syncthreads();
    v16h a[4], b[2];
#pragma unroll
    for (int mt = 0; mt < 4; ++mt)
      a[mt] = frag_a(&Xs[(waveM * 64 + mt * 16 + lane16) * LSTR + laneHalf * 8]);
#pragma unroll
    for (int nt = 0; nt < 2; ++nt)
      b[nt] = frag_b(&Ws[(waveN * 32 + nt * 16 + lane16) * LSTR + laneHalf * 8]);
#pragma unroll
    for (int mt = 0; mt < 4; ++mt)
#pragma unroll
      for (int nt = 0; nt < 2; ++nt) acc[mt][nt] = wmma16(a[mt], b[nt], acc[mt][nt]);
  }

  // ---- epilogue (C layout: lanes 0-15 M=vg, lanes 16-31 M=vg+8) ----
#pragma unroll
  for (int mt = 0; mt < 4; ++mt)
#pragma unroll
    for (int nt = 0; nt < 2; ++nt)
#pragma unroll
      for (int vg = 0; vg < 8; ++vg) {
        const int row = blockM + waveM * 64 + mt * 16 + laneHalf * 8 + vg;
        const int col = blockN + waveN * 32 + nt * 16 + lane16;
        float v = acc[mt][nt][vg];
        if (res) v += res[(size_t)row * O + col];
        out[(size_t)row * O + col] = v;
      }
}

// ---------------- RoPE + pack q/k/v to f16 [H][S][DH]; q pre-scaled 1/sqrt(DH) ----------------
__launch_bounds__(256)
__global__ void rope_pack(const float* __restrict__ q, const float* __restrict__ k,
                          const float* __restrict__ v, const float* __restrict__ cosT,
                          const float* __restrict__ sinT, const int* __restrict__ pos,
                          _Float16* __restrict__ qh, _Float16* __restrict__ kh,
                          _Float16* __restrict__ vh) {
  const int idx = blockIdx.x * 256 + threadIdx.x;  // over S*D
  const int dh = idx & (DHEAD - 1);
  const int h  = (idx >> 6) & (NHEAD - 1);
  const int s  = idx >> 10;
  const int p  = pos[s];
  const float c  = cosT[p * DHEAD + dh];
  const float sn = sinT[p * DHEAD + dh];
  const int base = s * DMODEL + h * DHEAD;
  const int pdh  = dh ^ 32;                         // rotate_half partner
  const float sgn = (dh < 32) ? -1.f : 1.f;
  const float qr = q[base + dh] * c + sgn * q[base + pdh] * sn;
  const float kr = k[base + dh] * c + sgn * k[base + pdh] * sn;
  const size_t ho = ((size_t)h * S_LEN + s) * DHEAD + dh;
  qh[ho] = (_Float16)(qr * 0.125f);
  kh[ho] = (_Float16)kr;
  vh[ho] = (_Float16)v[base + dh];
}

// ---------------- Flash attention: 64-query block, 4 waves x 16 rows ----------------
#define ASTR 72  // padded stride (144B)
__launch_bounds__(128)
__global__ void flash_attn(const _Float16* __restrict__ qh, const _Float16* __restrict__ kh,
                           const _Float16* __restrict__ vh, _Float16* __restrict__ oh) {
  const int h = blockIdx.y;
  const int qB = blockIdx.x * 64;
  const int t = threadIdx.x;
  const int wave = t >> 5, lane = t & 31, lane16 = lane & 15, laneHalf = lane >> 4;

  __shared__ _Float16 Qs[64 * ASTR];       // [qrow][dh]
  __shared__ _Float16 Ks[64 * ASTR];       // [kv][dh]
  __shared__ _Float16 Vt[64 * ASTR];       // [dh][kv]
  __shared__ _Float16 Pw[4 * 16 * ASTR];   // per-wave P scratch [16][64]

  {  // stage Q once (each thread: one 32-half segment)
    const int row = t >> 1, seg = (t & 1) * 32;
    const _Float16* src = qh + ((size_t)h * S_LEN + qB + row) * DHEAD + seg;
#pragma unroll
    for (int j = 0; j < 4; ++j)
      *(int4*)&Qs[row * ASTR + seg + j * 8] = ((const int4*)src)[j];
  }

  v8f oacc[4];
#pragma unroll
  for (int nd = 0; nd < 4; ++nd) oacc[nd] = (v8f){};
  float mrow[8], lrow[8];
#pragma unroll
  for (int vg = 0; vg < 8; ++vg) { mrow[vg] = -3e38f; lrow[vg] = 0.f; }

  for (int kvB = 0; kvB <= qB; kvB += 64) {
    __syncthreads();
    {  // stage K tile
      const int row = t >> 1, seg = (t & 1) * 32;
      const _Float16* src = kh + ((size_t)h * S_LEN + kvB + row) * DHEAD + seg;
#pragma unroll
      for (int j = 0; j < 4; ++j)
        *(int4*)&Ks[row * ASTR + seg + j * 8] = ((const int4*)src)[j];
    }
    {  // stage V tile transposed -> [dh][kv]
      const int kv = t >> 1, seg = (t & 1) * 32;
      const _Float16* src = vh + ((size_t)h * S_LEN + kvB + kv) * DHEAD + seg;
#pragma unroll
      for (int j = 0; j < 32; ++j) Vt[(seg + j) * ASTR + kv] = src[j];
    }
    __syncthreads();

    // scores S = Q K^T  (16 rows x 64 kv per wave)
    v8f sacc[4];
#pragma unroll
    for (int nt = 0; nt < 4; ++nt) sacc[nt] = (v8f){};
    const v16h a0 = frag_a(&Qs[(wave * 16 + lane16) * ASTR + laneHalf * 8]);
    const v16h a1 = frag_a(&Qs[(wave * 16 + lane16) * ASTR + 32 + laneHalf * 8]);
#pragma unroll
    for (int nt = 0; nt < 4; ++nt) {
      const v16h b0 = frag_b(&Ks[(nt * 16 + lane16) * ASTR + laneHalf * 8]);
      const v16h b1 = frag_b(&Ks[(nt * 16 + lane16) * ASTR + 32 + laneHalf * 8]);
      sacc[nt] = wmma16(a0, b0, sacc[nt]);
      sacc[nt] = wmma16(a1, b1, sacc[nt]);
    }
    if (kvB == qB) {  // causal mask on diagonal chunk
#pragma unroll
      for (int nt = 0; nt < 4; ++nt)
#pragma unroll
        for (int vg = 0; vg < 8; ++vg) {
          const int qr = qB + wave * 16 + laneHalf * 8 + vg;
          const int kv = kvB + nt * 16 + lane16;
          if (kv > qr) sacc[nt][vg] = -1e30f;
        }
    }
    // online softmax (row reductions across the 16-lane half)
#pragma unroll
    for (int vg = 0; vg < 8; ++vg) {
      float mx = sacc[0][vg];
#pragma unroll
      for (int nt = 1; nt < 4; ++nt) mx = fmaxf(mx, sacc[nt][vg]);
#pragma unroll
      for (int m = 1; m < 16; m <<= 1) mx = fmaxf(mx, __shfl_xor(mx, m, 32));
      const float mnew = fmaxf(mrow[vg], mx);
      const float alpha = __expf(mrow[vg] - mnew);
      mrow[vg] = mnew;
      float rs = 0.f;
#pragma unroll
      for (int nt = 0; nt < 4; ++nt) {
        const float pv = __expf(sacc[nt][vg] - mnew);
        sacc[nt][vg] = pv;
        rs += pv;
      }
#pragma unroll
      for (int m = 1; m < 16; m <<= 1) rs += __shfl_xor(rs, m, 32);
      lrow[vg] = lrow[vg] * alpha + rs;
#pragma unroll
      for (int nd = 0; nd < 4; ++nd) oacc[nd][vg] *= alpha;
    }
    // transpose P (C layout -> A layout) through per-wave LDS
#pragma unroll
    for (int nt = 0; nt < 4; ++nt)
#pragma unroll
      for (int vg = 0; vg < 8; ++vg)
        Pw[(wave * 16 + laneHalf * 8 + vg) * ASTR + nt * 16 + lane16] =
            (_Float16)sacc[nt][vg];
    const v16h p0 = frag_a(&Pw[(wave * 16 + lane16) * ASTR + laneHalf * 8]);
    const v16h p1 = frag_a(&Pw[(wave * 16 + lane16) * ASTR + 32 + laneHalf * 8]);
#pragma unroll
    for (int nd = 0; nd < 4; ++nd) {
      const v16h bv0 = frag_b(&Vt[(nd * 16 + lane16) * ASTR + laneHalf * 8]);
      const v16h bv1 = frag_b(&Vt[(nd * 16 + lane16) * ASTR + 32 + laneHalf * 8]);
      oacc[nd] = wmma16(p0, bv0, oacc[nd]);
      oacc[nd] = wmma16(p1, bv1, oacc[nd]);
    }
  }
  // epilogue: O/l  -> f16 [S][D] (head-interleaved)
#pragma unroll
  for (int nd = 0; nd < 4; ++nd)
#pragma unroll
    for (int vg = 0; vg < 8; ++vg) {
      const int row = qB + wave * 16 + laneHalf * 8 + vg;
      oh[(size_t)row * DMODEL + h * DHEAD + nd * 16 + lane16] =
          (_Float16)(oacc[nd][vg] / lrow[vg]);
    }
}

// ---------------- SwiGLU hadamard ----------------
__launch_bounds__(256)
__global__ void swiglu_f16(const float* __restrict__ gate, const float* __restrict__ up,
                           _Float16* __restrict__ had, size_t n) {
  const size_t i = (size_t)blockIdx.x * 256 + threadIdx.x;
  if (i < n) {
    const float g = gate[i];
    had[i] = (_Float16)(up[i] * (g / (1.f + __expf(-g))));
  }
}

// ---------------- host side ----------------
extern "C" void kernel_launch(void* const* d_in, const int* in_sizes, int n_in,
                              void* d_out, int out_size, void* d_ws, size_t ws_size,
                              hipStream_t stream) {
  (void)in_sizes; (void)n_in; (void)out_size; (void)ws_size;
  const float* x    = (const float*)d_in[0];
  const float* nw1  = (const float*)d_in[1];
  const float* nw2  = (const float*)d_in[2];
  const float* cosT = (const float*)d_in[3];
  const float* sinT = (const float*)d_in[4];
  const int*   pos  = (const int*)d_in[5];
  struct Proj { const int* codes; const float* absmax; const float* la; const float* lb; };
  auto P = [&](int b) {
    return Proj{(const int*)d_in[b], (const float*)d_in[b + 1],
                (const float*)d_in[b + 2], (const float*)d_in[b + 3]};
  };
  const Proj pq = P(6), pk = P(10), pv = P(14), po = P(18),
             pg = P(22), pu = P(26), pd = P(30);

  // workspace arena
  char* w = (char*)d_ws;
  size_t off = 0;
  auto take = [&](size_t bytes) -> void* {
    void* p = w + off;
    off += (bytes + 255) & ~(size_t)255;
    return p;
  };
  _Float16* h1h  = (_Float16*)take((size_t)S_LEN * DMODEL * 2);
  float*    q32  = (float*)take((size_t)S_LEN * DMODEL * 4);
  float*    k32  = (float*)take((size_t)S_LEN * DMODEL * 4);
  float*    v32  = (float*)take((size_t)S_LEN * DMODEL * 4);
  _Float16* qhB  = (_Float16*)take((size_t)S_LEN * DMODEL * 2);
  _Float16* khB  = (_Float16*)take((size_t)S_LEN * DMODEL * 2);
  _Float16* vhB  = (_Float16*)take((size_t)S_LEN * DMODEL * 2);
  _Float16* ohB  = (_Float16*)take((size_t)S_LEN * DMODEL * 2);
  float*    x1   = (float*)take((size_t)S_LEN * DMODEL * 4);
  _Float16* h2h  = (_Float16*)take((size_t)S_LEN * DMODEL * 2);
  float*    g32  = (float*)take((size_t)S_LEN * FFN * 4);
  float*    u32  = (float*)take((size_t)S_LEN * FFN * 4);
  _Float16* hadh = (_Float16*)take((size_t)S_LEN * FFN * 2);
  _Float16* xa   = (_Float16*)take((size_t)S_LEN * LRANK * 2);

  const dim3 gemmDD(DMODEL / 128, S_LEN / 128);
  const dim3 gemmDF(FFN / 128, S_LEN / 128);
  const size_t gemmLds = GEMM_LDS_BYTES;

  // h1 = rmsnorm(x, nw1)
  rmsnorm_f16<<<S_LEN, 256, 0, stream>>>(x, nw1, h1h, DMODEL);
  // q/k/v projections
  lora_xa<<<S_LEN, 256, 0, stream>>>(h1h, pq.la, xa, DMODEL);
  gemm_nf4_lora<<<gemmDD, 256, gemmLds, stream>>>(h1h, pq.codes, pq.absmax, xa, pq.lb,
                                                  nullptr, q32, DMODEL, DMODEL);
  lora_xa<<<S_LEN, 256, 0, stream>>>(h1h, pk.la, xa, DMODEL);
  gemm_nf4_lora<<<gemmDD, 256, gemmLds, stream>>>(h1h, pk.codes, pk.absmax, xa, pk.lb,
                                                  nullptr, k32, DMODEL, DMODEL);
  lora_xa<<<S_LEN, 256, 0, stream>>>(h1h, pv.la, xa, DMODEL);
  gemm_nf4_lora<<<gemmDD, 256, gemmLds, stream>>>(h1h, pv.codes, pv.absmax, xa, pv.lb,
                                                  nullptr, v32, DMODEL, DMODEL);
  // rope + pack
  rope_pack<<<(S_LEN * DMODEL) / 256, 256, 0, stream>>>(q32, k32, v32, cosT, sinT,
                                                        pos, qhB, khB, vhB);
  // causal flash attention
  flash_attn<<<dim3(S_LEN / 64, NHEAD), 128, 0, stream>>>(qhB, khB, vhB, ohB);
  // o-projection + first residual  (x1 = x + o W_o^T + lora)
  lora_xa<<<S_LEN, 256, 0, stream>>>(ohB, po.la, xa, DMODEL);
  gemm_nf4_lora<<<gemmDD, 256, gemmLds, stream>>>(ohB, po.codes, po.absmax, xa, po.lb,
                                                  x, x1, DMODEL, DMODEL);
  // MLP
  rmsnorm_f16<<<S_LEN, 256, 0, stream>>>(x1, nw2, h2h, DMODEL);
  lora_xa<<<S_LEN, 256, 0, stream>>>(h2h, pg.la, xa, DMODEL);
  gemm_nf4_lora<<<gemmDF, 256, gemmLds, stream>>>(h2h, pg.codes, pg.absmax, xa, pg.lb,
                                                  nullptr, g32, DMODEL, FFN);
  lora_xa<<<S_LEN, 256, 0, stream>>>(h2h, pu.la, xa, DMODEL);
  gemm_nf4_lora<<<gemmDF, 256, gemmLds, stream>>>(h2h, pu.codes, pu.absmax, xa, pu.lb,
                                                  nullptr, u32, DMODEL, FFN);
  swiglu_f16<<<(S_LEN * FFN) / 256, 256, 0, stream>>>(g32, u32, hadh,
                                                      (size_t)S_LEN * FFN);
  // down projection + second residual straight into d_out
  lora_xa<<<S_LEN, 256, 0, stream>>>(hadh, pd.la, xa, FFN);
  gemm_nf4_lora<<<gemmDD, 256, gemmLds, stream>>>(hadh, pd.codes, pd.absmax, xa, pd.lb,
                                                  x1, (float*)d_out, FFN, DMODEL);
}
